// MambaBlockFallback_15401752723576
// MI455X (gfx1250) — matrix operand order repaired and checked by reading, not compiled, not run on hardware
//
#include <hip/hip_runtime.h>
#include <hip/hip_bf16.h>
#include <math.h>

// ---------------------------------------------------------------------------
// Mamba block for MI455X (gfx1250, wave32).
// GEMMs: v_wmma_f32_16x16x32_bf16, 64x128 block tile, 8 waves x (16x64 = 4
// accumulators). Weights are pre-packed once per launch into bf16-pair K-tiles
// in the exact LDS layout, so B staging is a raw 8KB global->LDS copy done
// with global_load_async_to_lds_b128 (ASYNCcnt path, no VGPR round-trip),
// fenced with s_wait_asynccnt before the tile barrier. A (activations) is
// converted f32->bf16 in staging via v_cvt_pk_bf16_f32. LDS double-buffered
// (24KB of the 320KB WGP pool). Scan keeps h in registers (1 f32/lane = one
// (channel,state) pair) across 512 blocks to hide L2 latency on the
// sequential critical path. Workspace: ~190 MB.
// ---------------------------------------------------------------------------

typedef __bf16 bf16;
typedef __attribute__((ext_vector_type(16))) __bf16        v16bf;
typedef __attribute__((ext_vector_type(8)))  float         v8f;
typedef __attribute__((ext_vector_type(8)))  unsigned int  v8u;

#define B_SZ    2
#define L_SZ    2048
#define D_MODEL 1024
#define D_STATE 16
#define D_CONV  4
#define D_INNER 2048
#define N_XZ    (2 * D_INNER)            // 4096
#define N_PROJ  (2 * D_STATE + D_INNER)  // 2080
#define M_ROWS  (B_SZ * L_SZ)            // 4096

#define TM 64
#define TN 128
#define TK 32
#define TK2 16        // TK/2 packed bf16 pairs

#if defined(__has_builtin)
#if __has_builtin(__builtin_amdgcn_global_load_async_to_lds_b128)
#define HAVE_ASYNC_B128 1
#endif
#endif
#ifndef HAVE_ASYNC_B128
#define HAVE_ASYNC_B128 0
#endif

#if HAVE_ASYNC_B128
// exact parameter types per hipcc diagnostic: <4 x i32>* in AS1 / AS3
typedef __attribute__((ext_vector_type(4))) int v4i_async;
typedef __attribute__((address_space(1))) v4i_async* g_v4i_ptr;
typedef __attribute__((address_space(3))) v4i_async* l_v4i_ptr;
#endif

__device__ __forceinline__ float siluf(float x) {
  return x * (1.0f / (1.0f + __expf(-x)));
}
__device__ __forceinline__ float softplusf_(float x) {
  return (x > 20.0f) ? x : log1pf(__expf(x));
}
// Force a single v_cvt_pk_bf16_f32 (RNE) per pair — mnemonic/operand form
// verified in earlier gfx1250 disassembly; avoids the compiler's half-insert
// repair sequences from the union-based pattern.
__device__ __forceinline__ unsigned int pk2bf(float lo, float hi) {
  unsigned int r;
  asm("v_cvt_pk_bf16_f32 %0, %1, %2" : "=v"(r) : "v"(lo), "v"(hi));
  return r;
}
__device__ __forceinline__ void async_fence() {
#if HAVE_ASYNC_B128
#if defined(__has_builtin) && __has_builtin(__builtin_amdgcn_s_wait_asynccnt)
  __builtin_amdgcn_s_wait_asynccnt(0);
#else
  asm volatile("s_wait_asynccnt 0x0" ::: "memory");
#endif
#endif
}

// ---------------------------------------------------------------------------
// One-shot weight pre-pack: B[K][N] f32 (ldb == N) -> Bp[kTile][Npad][16] u32,
// each u32 = packed bf16 pair (k even, k odd); columns >= N zero-filled so the
// GEMM needs no B-side edge handling. One thread per (kTile, col): 32 coalesced
// row loads, 64B contiguous store.
// ---------------------------------------------------------------------------
__global__ __launch_bounds__(256) void pack_b_kernel(
    const float* __restrict__ B, unsigned* __restrict__ Bp,
    int K, int N, int Npad)
{
  int i = blockIdx.x * 256 + threadIdx.x;
  int nK = K / TK;
  if (i >= nK * Npad) return;
  int c  = i % Npad;
  int kt = i / Npad;
  const float* src = B + (size_t)(kt * TK) * N + c;
  unsigned out[16];
  #pragma unroll
  for (int q = 0; q < 16; ++q) {
    float lo = 0.f, hi = 0.f;
    if (c < N) {
      lo = src[(size_t)(2 * q) * N];
      hi = src[(size_t)(2 * q + 1) * N];
    }
    out[q] = pk2bf(lo, hi);
  }
  unsigned* dst = Bp + (size_t)i * 16;
  *(uint4*)(dst)      = *(const uint4*)(out);
  *(uint4*)(dst + 4)  = *(const uint4*)(out + 4);
  *(uint4*)(dst + 8)  = *(const uint4*)(out + 8);
  *(uint4*)(dst + 12) = *(const uint4*)(out + 12);
}

// ---------------------------------------------------------------------------
// C[M,N] = act(A[M,K] @ B[K,N] + bias), B pre-packed as above.
// Requires M % 64 == 0, K % 32 == 0 (true for all launches here).
// act: 0 = none, 1 = softplus.
// ---------------------------------------------------------------------------
__global__ __launch_bounds__(256) void gemm_bf16_wmma(
    const float* __restrict__ A, const unsigned* __restrict__ Bp,
    float* __restrict__ C,
    int M, int N, int K, int lda, int Npad, int ldc,
    const float* __restrict__ bias, int act)
{
  // Packed bf16-pair tiles, double buffered: 2*(4KB + 8KB) = 24KB.
  __shared__ unsigned As32[2][TM][TK2];
  __shared__ unsigned Bs[2][TN][TK2];

  const int tid  = threadIdx.x;
  const int wid  = tid >> 5;           // 0..7
  const int lane = tid & 31;
  const int half = lane >> 4;          // selects K sub-range per ISA layout
  const int lm   = lane & 15;

  const int rowBase = blockIdx.y * TM;
  const int colBase = blockIdx.x * TN;
  const int row0 = (wid & 3) * 16;     // 4 row groups
  const int col0 = (wid >> 2) * 64;    // 2 col groups, 4 tiles of 16 each

  // ---- per-thread staging state (computed once, advanced incrementally) --
  const int  ar_st = tid >> 2;                 // A row
  const int  ak_st = (tid & 3) * 8;            // A first K of 8
  const float* aPtr = A + (size_t)(rowBase + ar_st) * lda + ak_st;

  // B tile = one contiguous 8KB chunk per kTile: thread copies 32B.
  const unsigned* bPtr = Bp + ((size_t)colBase) * 16 + (size_t)tid * 8;
  const size_t bStep = (size_t)Npad * 16;      // advance per K-step
  unsigned* const bDst0 = &Bs[0][0][0] + tid * 8;

  v8f acc0 = {}, acc1 = {}, acc2 = {}, acc3 = {};

  auto stage = [&](int p) {
    // ---- A: two float4 loads -> 4 packs -> one ds_store_b128 -------------
    {
      float4 v0 = *(const float4*)aPtr;
      float4 v1 = *(const float4*)(aPtr + 4);
      uint4 w;
      w.x = pk2bf(v0.x, v0.y); w.y = pk2bf(v0.z, v0.w);
      w.z = pk2bf(v1.x, v1.y); w.w = pk2bf(v1.z, v1.w);
      *(uint4*)&As32[p][ar_st][ak_st >> 1] = w;
    }
    // ---- B: raw 32B async copy of pre-packed tile ------------------------
    {
      unsigned* dst = bDst0 + (size_t)p * (TN * TK2);
#if HAVE_ASYNC_B128
      __builtin_amdgcn_global_load_async_to_lds_b128(
          (g_v4i_ptr)bPtr,       (l_v4i_ptr)dst,       0, 0);
      __builtin_amdgcn_global_load_async_to_lds_b128(
          (g_v4i_ptr)(bPtr + 4), (l_v4i_ptr)(dst + 4), 0, 0);
#else
      uint4 v0 = *(const uint4*)bPtr;
      uint4 v1 = *(const uint4*)(bPtr + 4);
      *(uint4*)dst       = v0;
      *(uint4*)(dst + 4) = v1;
#endif
    }
  };

  // ---- main loop: double buffered, one barrier per K-step ----------------
  const int nK = K / TK;
  stage(0);
  aPtr += TK; bPtr += bStep;
  async_fence();
  __syncthreads();

  for (int kt = 0; kt < nK; ++kt) {
    const int p = kt & 1;
    if (kt + 1 < nK) {                 // overlap next-tile loads with WMMAs
      stage(p ^ 1);
      aPtr += TK; bPtr += bStep;
    }

    // A fragment: lane half selects K {0..7,16..23} / {8..15,24..31};
    // packed pairs -> two 16B-aligned ds_load_b128.
    v8u au;
    {
      int arr = row0 + lm;
      int ka  = half * 4;
      uint4 a0 = *(const uint4*)&As32[p][arr][ka];
      uint4 a1 = *(const uint4*)&As32[p][arr][ka + 8];
      au[0] = a0.x; au[1] = a0.y; au[2] = a0.z; au[3] = a0.w;
      au[4] = a1.x; au[5] = a1.y; au[6] = a1.z; au[7] = a1.w;
    }
    v16bf af = __builtin_bit_cast(v16bf, au);

    // 4 B fragments (col tiles) reuse the A fragment: 4 WMMAs per K-step.
    #pragma unroll
    for (int n = 0; n < 4; ++n) {
      int bc = col0 + n * 16 + lm;
      uint4 b0 = *(const uint4*)&Bs[p][bc][half * 4];
      uint4 b1 = *(const uint4*)&Bs[p][bc][half * 4 + 8];
      v8u bu;
      bu[0] = b0.x; bu[1] = b0.y; bu[2] = b0.z; bu[3] = b0.w;
      bu[4] = b1.x; bu[5] = b1.y; bu[6] = b1.z; bu[7] = b1.w;
      v16bf bf_ = __builtin_bit_cast(v16bf, bu);
      v8f* accp = (n == 0) ? &acc0 : (n == 1) ? &acc1 : (n == 2) ? &acc2 : &acc3;
      *accp = __builtin_amdgcn_wmma_f32_16x16x32_bf16(
          false, af, false, bf_, (short)0, *accp, false, false);
    }
    async_fence();
    __syncthreads();
  }

  // ---- epilogue: C/D layout VGPR v -> M = v + 8*half, N = lane%16 --------
  #pragma unroll
  for (int n = 0; n < 4; ++n) {
    const v8f& acc = (n == 0) ? acc0 : (n == 1) ? acc1 : (n == 2) ? acc2 : acc3;
    int c = colBase + col0 + n * 16 + lm;
    if (c < N) {
      float bs = bias ? bias[c] : 0.0f;
      #pragma unroll
      for (int v = 0; v < 8; ++v) {
        int r = rowBase + row0 + half * 8 + v;   // r < M by construction
        float val = acc[v] + bs;
        if (act == 1) val = softplusf_(val);
        C[(size_t)r * ldc + c] = val;
      }
    }
  }
}

// ---------------------------------------------------------------------------
// Depthwise causal conv (width 4) + SiLU; xb is the low half of xz.
// ---------------------------------------------------------------------------
__global__ __launch_bounds__(256) void conv_silu_kernel(
    const float* __restrict__ xz, const float* __restrict__ cw,
    const float* __restrict__ cb, float* __restrict__ u)
{
  long long i = (long long)blockIdx.x * blockDim.x + threadIdx.x;
  if (i >= (long long)B_SZ * L_SZ * D_INNER) return;
  int d = (int)(i % D_INNER);
  int l = (int)((i / D_INNER) % L_SZ);
  int b = (int)(i / ((long long)D_INNER * L_SZ));
  float acc = cb[d];
  #pragma unroll
  for (int j = 0; j < D_CONV; ++j) {
    int t = l - (D_CONV - 1) + j;
    if (t >= 0)
      acc = __fmaf_rn(xz[((long long)b * L_SZ + t) * N_XZ + d], cw[d * D_CONV + j], acc);
  }
  u[i] = siluf(acc);
}

// ---------------------------------------------------------------------------
// Selective scan: one lane = one (channel, state); h register-resident.
// 16-lane state reduction via __shfl_xor 1/2/4/8 (stays in aligned 16-groups
// on wave32). y fuses (+u*D)*silu(z) and safely reuses the dt buffer.
// ---------------------------------------------------------------------------
__global__ __launch_bounds__(128) void mamba_scan(
    const float* __restrict__ proj,   // (B,L,2080): Bm=[0:16], Cm=[16:32]
    const float* __restrict__ dt,     // (B,L,2048) softplus'ed
    const float* __restrict__ u,      // (B,L,2048)
    const float* __restrict__ xz,     // (B,L,4096): z at +2048
    const float* __restrict__ h_init, // (B,2048,16)
    const float* __restrict__ A_log,  // (2048,16)
    const float* __restrict__ D_param,// (2048,)
    float* __restrict__ y,            // (B,L,2048) -- aliases dt buffer
    float* __restrict__ h_out)        // (B,2048,16)
{
  const int blocksPerBatch = D_INNER / 8;      // 256
  const int b    = blockIdx.x / blocksPerBatch;
  const int cblk = blockIdx.x % blocksPerBatch;
  const int tid  = threadIdx.x;
  const int s    = tid & 15;
  const int c    = cblk * 8 + (tid >> 4);

  const float Ac = -__expf(A_log[c * D_STATE + s]);
  const float Dc = D_param[c];
  float h = h_init[((long long)b * D_INNER + c) * D_STATE + s];

  const float* pj  = proj + (long long)b * L_SZ * N_PROJ;
  const float* dtp = dt   + (long long)b * L_SZ * D_INNER;
  const float* up  = u    + (long long)b * L_SZ * D_INNER;
  const float* zp  = xz   + (long long)b * L_SZ * N_XZ + D_INNER;
  float*       yp  = y    + (long long)b * L_SZ * D_INNER;

  for (int t = 0; t < L_SZ; ++t) {
    const long long pbase = (long long)t * N_PROJ;
    float bt  = pj[pbase + s];
    float ct  = pj[pbase + D_STATE + s];
    float dtt = dtp[(long long)t * D_INNER + c];
    float ut  = up [(long long)t * D_INNER + c];

    float dA = __expf(dtt * Ac);
    h = __fmaf_rn(dA, h, dtt * bt * ut);   // h = dA*h + dt*B*u

    float part = h * ct;                   // y_t = sum_s h*C
    part += __shfl_xor(part, 1);
    part += __shfl_xor(part, 2);
    part += __shfl_xor(part, 4);
    part += __shfl_xor(part, 8);

    if (s == 0) {
      float zt = zp[(long long)t * N_XZ + c];
      yp[(long long)t * D_INNER + c] = (part + ut * Dc) * siluf(zt);
    }
  }
  h_out[((long long)b * D_INNER + c) * D_STATE + s] = h;
}

// ---------------------------------------------------------------------------
extern "C" void kernel_launch(void* const* d_in, const int* in_sizes, int n_in,
                              void* d_out, int out_size, void* d_ws, size_t ws_size,
                              hipStream_t stream) {
  const float* x       = (const float*)d_in[0];
  const float* h_init  = (const float*)d_in[1];
  const float* W_in    = (const float*)d_in[2];
  const float* conv_w  = (const float*)d_in[3];
  const float* conv_b  = (const float*)d_in[4];
  const float* W_x     = (const float*)d_in[5];
  const float* W_dt    = (const float*)d_in[6];
  const float* b_dt    = (const float*)d_in[7];
  const float* A_log   = (const float*)d_in[8];
  const float* D_param = (const float*)d_in[9];
  const float* W_out   = (const float*)d_in[10];

  float* out    = (float*)d_out;                           // (B,L,D_MODEL)
  float* h_last = out + (size_t)B_SZ * L_SZ * D_MODEL;     // (B,D_INNER,16)

  // Workspace: xz | u | proj | dt(=y) | packed weights  ~= 190 MB total.
  float* xz   = (float*)d_ws;
  float* u    = xz   + (size_t)B_SZ * L_SZ * N_XZ;
  float* proj = u    + (size_t)B_SZ * L_SZ * D_INNER;
  float* dtb  = proj + (size_t)B_SZ * L_SZ * N_PROJ;       // reused as y

  const int NPAD_IN  = 4096;                 // N_XZ
  const int NPAD_X   = 2176;                 // ceil(2080/128)*128
  const int NPAD_DT  = 2048;
  const int NPAD_OUT = 1024;
  unsigned* bWin  = (unsigned*)(dtb + (size_t)B_SZ * L_SZ * D_INNER);
  unsigned* bWx   = bWin + (size_t)(D_MODEL / TK) * NPAD_IN  * 16;
  unsigned* bWdt  = bWx  + (size_t)(D_INNER / TK) * NPAD_X   * 16;
  unsigned* bWout = bWdt + (size_t)(D_INNER / TK) * NPAD_DT  * 16;

  dim3 blk(256);

  // 0) pre-pack weights to bf16-pair tiles (one shot, streaming)
  pack_b_kernel<<<dim3(((D_MODEL / TK) * NPAD_IN  + 255) / 256), blk, 0, stream>>>(
      W_in, bWin, D_MODEL, N_XZ, NPAD_IN);
  pack_b_kernel<<<dim3(((D_INNER / TK) * NPAD_X   + 255) / 256), blk, 0, stream>>>(
      W_x, bWx, D_INNER, N_PROJ, NPAD_X);
  pack_b_kernel<<<dim3(((D_INNER / TK) * NPAD_DT  + 255) / 256), blk, 0, stream>>>(
      W_dt, bWdt, D_INNER, D_INNER, NPAD_DT);
  pack_b_kernel<<<dim3(((D_INNER / TK) * NPAD_OUT + 255) / 256), blk, 0, stream>>>(
      W_out, bWout, D_INNER, D_MODEL, NPAD_OUT);

  // 1) xz = x @ W_in                         (M=4096, N=4096, K=1024)
  gemm_bf16_wmma<<<dim3(N_XZ / TN, M_ROWS / TM), blk, 0, stream>>>(
      x, bWin, xz, M_ROWS, N_XZ, D_MODEL, D_MODEL, NPAD_IN, N_XZ, nullptr, 0);

  // 2) u = silu(causal_depthwise_conv(xb) + conv_b)
  conv_silu_kernel<<<dim3((B_SZ * L_SZ * D_INNER) / 256), blk, 0, stream>>>(
      xz, conv_w, conv_b, u);

  // 3) proj = u @ W_x                        (M=4096, N=2080, K=2048)
  gemm_bf16_wmma<<<dim3(NPAD_X / TN, M_ROWS / TM), blk, 0, stream>>>(
      u, bWx, proj, M_ROWS, N_PROJ, D_INNER, D_INNER, NPAD_X, N_PROJ, nullptr, 0);

  // 4) dt = softplus(proj[...,32:] @ W_dt + b_dt)   (M=4096, N=2048, K=2048)
  gemm_bf16_wmma<<<dim3(NPAD_DT / TN, M_ROWS / TM), blk, 0, stream>>>(
      proj + 2 * D_STATE, bWdt, dtb, M_ROWS, D_INNER, D_INNER,
      N_PROJ, NPAD_DT, D_INNER, b_dt, 1);

  // 5) sequential selective scan + gated epilogue; writes y (over dtb), h_last
  mamba_scan<<<dim3(B_SZ * (D_INNER / 8)), dim3(128), 0, stream>>>(
      proj, dtb, u, xz, h_init, A_log, D_param, dtb, h_last);

  // 6) out = y @ W_out                       (M=4096, N=1024, K=2048)
  gemm_bf16_wmma<<<dim3(NPAD_OUT / TN, M_ROWS / TM), blk, 0, stream>>>(
      dtb, bWout, out, M_ROWS, D_MODEL, D_INNER, D_INNER, NPAD_OUT, D_MODEL,
      nullptr, 0);
}
